// AffineCouplingLayer_13932873909153
// MI455X (gfx1250) — matrix operand order repaired and checked by reading, not compiled, not run on hardware
//
#include <hip/hip_runtime.h>
#include <hip/hip_bf16.h>
#include <stdint.h>

typedef __bf16 bf16_t;
typedef __attribute__((ext_vector_type(16))) __bf16 v16bf;
typedef __attribute__((ext_vector_type(8)))  float   v8f;
typedef __attribute__((ext_vector_type(4)))  unsigned int v4u;
typedef __attribute__((ext_vector_type(8)))  int v8i;
typedef __attribute__((ext_vector_type(4)))  int v4i;

// ---------------- problem sizes ----------------
constexpr int B_     = 65536;
constexpr int D_     = 256;
constexpr int C_     = 128;
constexpr int H_     = 1024;
constexpr int SPLIT_ = 128;   // D/2
constexpr int NETIN_ = 256;   // D - SPLIT + C

// ---------------- GEMM tiling ------------------
constexpr int BM = 128, BN = 128, BK = 32;
constexpr int LDSK = BK + 8;          // padded row stride (16 DWORDs data + 4 DWORDs pad)
constexpr int WAVES_M = 4, WAVES_N = 2;
constexpr int WM = BM / WAVES_M;      // 32 rows / wave
constexpr int WN = BN / WAVES_N;      // 64 cols / wave
constexpr int MF = WM / 16;           // 2 m-fragments
constexpr int NF = WN / 16;           // 4 n-fragments

// ---- CDNA5 async global->LDS copy (ASYNCcnt tracked) ----
__device__ __forceinline__ void async_ld16(void* lds_ptr, const void* gptr) {
  uint32_t lds_addr = (uint32_t)(uintptr_t)lds_ptr;   // LDS aperture: low 32 bits
  asm volatile("global_load_async_to_lds_b128 %0, %1, off"
               :: "v"(lds_addr), "v"(gptr)
               : "memory");
}
__device__ __forceinline__ void wait_async0() {
  asm volatile("s_wait_asynccnt 0x0" ::: "memory");
}

// ---- CDNA5 Tensor Data Mover: one-instruction 2D tile load into padded LDS ----
// D# per CDNA5 ISA: group0 {count,lds_addr,global_addr,type=2},
// group1 {data_size=2B, pad 16dw interval / 4dw amount, dims, strides}.
__device__ __forceinline__ void tdm_load_tile(void* lds_ptr, const void* gptr,
                                              uint32_t tensor_d0, uint32_t tensor_d1,
                                              uint32_t tile_d0, uint32_t tile_d1,
                                              uint32_t stride_elems) {
  uint64_t ga = (uint64_t)(uintptr_t)gptr;
  v4u g0;
  g0[0] = 1u;                                              // count=1 (valid user D#)
  g0[1] = (uint32_t)(uintptr_t)lds_ptr;                    // lds_addr
  g0[2] = (uint32_t)(ga & 0xffffffffu);                    // global_addr[31:0]
  g0[3] = (uint32_t)((ga >> 32) & 0x1ffffffu) | (2u << 30);// global_addr[56:32] | type=2
  v8i g1;
  g1[0] = (int)((1u << 16)      // data_size = 2 bytes (bf16)
              | (1u << 20)      // pad_enable
              | (3u << 22)      // pad_interval: 16 DWORDs (=tile row 64B)
              | (3u << 25));    // pad_amount:   4 DWORDs (=16B) -> LDS row stride 80B
  g1[1] = (int)((tensor_d0 & 0xffffu) << 16);                              // dim0[15:0]
  g1[2] = (int)(((tensor_d0 >> 16) & 0xffffu) | ((tensor_d1 & 0xffffu) << 16));
  g1[3] = (int)(((tensor_d1 >> 16) & 0xffffu) | ((tile_d0 & 0xffffu) << 16));
  g1[4] = (int)(tile_d1 & 0xffffu);                                        // tile_dim1
  g1[5] = (int)stride_elems;                                               // dim0_stride
  g1[6] = 0;
  g1[7] = 0;
  v4i g2 = {0, 0, 0, 0};
  v4i g3 = {0, 0, 0, 0};
#if __clang_major__ >= 23
  v8i g4 = {0, 0, 0, 0, 0, 0, 0, 0};
  __builtin_amdgcn_tensor_load_to_lds(g0, g1, g2, g3, g4, 0);
#else
  __builtin_amdgcn_tensor_load_to_lds(g0, g1, g2, g3, 0);
#endif
}

// A-matrix 16x32 bf16 fragment (CDNA5 ISA 7.12.2 layout)
__device__ __forceinline__ v16bf ld_fragA(const bf16_t* p) {
  union { v16bf v; uint4 q[2]; } u;
  u.q[0] = *(const uint4*)(p);
  u.q[1] = *(const uint4*)(p + 16);
  return u.v;
}
// B-matrix 32x16 fragment from an [N][K] tile
__device__ __forceinline__ v16bf ld_fragB(const bf16_t* p) {
  union { v16bf v; uint4 q[2]; } u;
  u.q[0] = *(const uint4*)(p);
  u.q[1] = *(const uint4*)(p + 16);
  return u.v;
}

__device__ __forceinline__ v8f wmma_bf16(v16bf a, v16bf b, v8f c) {
  return __builtin_amdgcn_wmma_f32_16x16x32_bf16(false, a, false, b, (short)0, c,
                                                 false, false);
}

// ---------------- helper kernels ----------------
__global__ void pack_a0(const float* __restrict__ x, const float* __restrict__ cond,
                        bf16_t* __restrict__ A0) {
  size_t i = (size_t)blockIdx.x * 256 + threadIdx.x;
  int b = (int)(i >> 8);
  int k = (int)(i & 255);
  float v = (k < SPLIT_) ? x[(size_t)b * D_ + 2 * k]
                         : cond[(size_t)b * C_ + (k - SPLIT_)];
  A0[i] = (bf16_t)v;
}

__global__ void convert_transpose(const float* __restrict__ W, bf16_t* __restrict__ Wt,
                                  int K, int N) {
  size_t i = (size_t)blockIdx.x * 256 + threadIdx.x;
  if (i >= (size_t)K * N) return;
  int n = (int)(i / K);
  int k = (int)(i % K);
  Wt[i] = (bf16_t)W[(size_t)k * N + n];
}

// ---------------- main GEMM: Out = relu(A @ W + bias), TDM-fed ----------------
__global__ __launch_bounds__(256, 1)
void gemm_bf16_relu(const bf16_t* __restrict__ A,   // [M][Ktot]
                    const bf16_t* __restrict__ Bt,  // [Ntot][Ktot]
                    const float*  __restrict__ bias,
                    bf16_t* __restrict__ Out,       // [M][Ntot]
                    int Ktot, int Ntot) {
  __shared__ bf16_t lA[2][BM * LDSK];
  __shared__ bf16_t lB[2][BN * LDSK];

  const int tid  = threadIdx.x;
  const int lane = tid & 31;
  const int wid  = tid >> 5;
  const int wm   = wid >> 1;
  const int wn   = wid & 1;
  const int r16  = lane & 15;
  const int hA   = (lane >> 4) << 3;
  const int hB   = (lane >> 4) << 3;
  const int mb   = blockIdx.y;
  const int nb   = blockIdx.x;

  // one wave drives the Tensor Data Mover for both tiles (EXEC-independent DMA)
  auto load_tiles_tdm = [&](int buf, int k0) {
    if (wid == 0) {
      tdm_load_tile(&lA[buf][0], A  + (size_t)(mb * BM) * Ktot + k0,
                    (uint32_t)Ktot, BM, BK, BM, (uint32_t)Ktot);
      tdm_load_tile(&lB[buf][0], Bt + (size_t)(nb * BN) * Ktot + k0,
                    (uint32_t)Ktot, BN, BK, BN, (uint32_t)Ktot);
    }
  };

  v8f zero = {0.f, 0.f, 0.f, 0.f, 0.f, 0.f, 0.f, 0.f};
  v8f acc[MF][NF];
#pragma unroll
  for (int i = 0; i < MF; ++i)
#pragma unroll
    for (int j = 0; j < NF; ++j) acc[i][j] = zero;

  load_tiles_tdm(0, 0);
  if (wid == 0) __builtin_amdgcn_s_wait_tensorcnt(0);
  __syncthreads();

  const int nk = Ktot / BK;
  for (int kt = 0; kt < nk; ++kt) {
    const int buf = kt & 1;
    if (kt + 1 < nk) load_tiles_tdm(buf ^ 1, (kt + 1) * BK);   // overlapped with WMMAs
    if (kt + 2 < nk) {
      __builtin_prefetch(A  + (size_t)(mb * BM + (tid & 127)) * Ktot + (kt + 2) * BK, 0, 0);
      __builtin_prefetch(Bt + (size_t)(nb * BN + (tid & 127)) * Ktot + (kt + 2) * BK, 0, 0);
    }
    // preload all fragments into distinct registers, then issue 8 independent WMMAs
    v16bf af[MF], bfv[NF];
#pragma unroll
    for (int mf = 0; mf < MF; ++mf)
      af[mf] = ld_fragA(&lA[buf][(wm * WM + mf * 16 + r16) * LDSK + hA]);
#pragma unroll
    for (int nf = 0; nf < NF; ++nf)
      bfv[nf] = ld_fragB(&lB[buf][(wn * WN + nf * 16 + r16) * LDSK + hB]);
#pragma unroll
    for (int nf = 0; nf < NF; ++nf)
#pragma unroll
      for (int mf = 0; mf < MF; ++mf)
        acc[mf][nf] = wmma_bf16(af[mf], bfv[nf], acc[mf][nf]);
    if (wid == 0) __builtin_amdgcn_s_wait_tensorcnt(0);
    __syncthreads();
  }

#pragma unroll
  for (int mf = 0; mf < MF; ++mf) {
    const int row0 = mb * BM + wm * WM + mf * 16 + ((lane >> 4) << 3);
#pragma unroll
    for (int nf = 0; nf < NF; ++nf) {
      const int col = nb * BN + wn * WN + nf * 16 + r16;
      const float bv = bias[col];
#pragma unroll
      for (int r = 0; r < 8; ++r) {
        float v = acc[mf][nf][r] + bv;
        v = v > 0.f ? v : 0.f;
        Out[(size_t)(row0 + r) * Ntot + col] = (bf16_t)v;
      }
    }
  }
}

// ---------------- GEMM3 + affine epilogue (async-load fed) ----------------
__global__ __launch_bounds__(256, 1)
void gemm3_affine(const bf16_t* __restrict__ Hin,  // [B][H]
                  const bf16_t* __restrict__ Wst,  // [SPLIT][H]
                  const bf16_t* __restrict__ Wtt,  // [SPLIT][H]
                  const float* __restrict__ bs, const float* __restrict__ bt,
                  const float* __restrict__ x,     // [B][D]
                  float* __restrict__ out,         // [B][D]
                  float* __restrict__ logdet) {    // [B]
  __shared__ bf16_t lA[2][BM * LDSK];
  __shared__ bf16_t lS[2][BN * LDSK];
  __shared__ bf16_t lT[2][BN * LDSK];
  __shared__ float  red[BM][WAVES_N];

  const int tid  = threadIdx.x;
  const int lane = tid & 31;
  const int wid  = tid >> 5;
  const int wm   = wid >> 1;
  const int wn   = wid & 1;
  const int r16  = lane & 15;
  const int hA   = (lane >> 4) << 3;
  const int hB   = (lane >> 4) << 3;
  const int mb   = blockIdx.x;

  auto load_tiles_async = [&](int buf, int k0) {
#pragma unroll
    for (int p = 0; p < 2; ++p) {
      int i = p * 256 + tid;
      int row = i >> 2, seg = i & 3;
      async_ld16(&lA[buf][row * LDSK + seg * 8],
                 Hin + (size_t)(mb * BM + row) * H_ + k0 + seg * 8);
      async_ld16(&lS[buf][row * LDSK + seg * 8],
                 Wst + (size_t)row * H_ + k0 + seg * 8);
      async_ld16(&lT[buf][row * LDSK + seg * 8],
                 Wtt + (size_t)row * H_ + k0 + seg * 8);
    }
  };

  v8f zero = {0.f, 0.f, 0.f, 0.f, 0.f, 0.f, 0.f, 0.f};
  v8f acc_s[MF][NF], acc_t[MF][NF];
#pragma unroll
  for (int i = 0; i < MF; ++i)
#pragma unroll
    for (int j = 0; j < NF; ++j) { acc_s[i][j] = zero; acc_t[i][j] = zero; }

  load_tiles_async(0, 0);
  wait_async0();
  __syncthreads();

  const int nk = H_ / BK;   // 32
  for (int kt = 0; kt < nk; ++kt) {
    const int buf = kt & 1;
    if (kt + 1 < nk) load_tiles_async(buf ^ 1, (kt + 1) * BK);
    if (kt + 2 < nk)
      __builtin_prefetch(Hin + (size_t)(mb * BM + (tid & 127)) * H_ + (kt + 2) * BK, 0, 0);
    v16bf af[MF];
#pragma unroll
    for (int mf = 0; mf < MF; ++mf)
      af[mf] = ld_fragA(&lA[buf][(wm * WM + mf * 16 + r16) * LDSK + hA]);
#pragma unroll
    for (int nf = 0; nf < NF; ++nf) {
      v16bf sf = ld_fragB(&lS[buf][(wn * WN + nf * 16 + r16) * LDSK + hB]);
      v16bf tf = ld_fragB(&lT[buf][(wn * WN + nf * 16 + r16) * LDSK + hB]);
#pragma unroll
      for (int mf = 0; mf < MF; ++mf) {
        acc_s[mf][nf] = wmma_bf16(af[mf], sf, acc_s[mf][nf]);
        acc_t[mf][nf] = wmma_bf16(af[mf], tf, acc_t[mf][nf]);
      }
    }
    wait_async0();
    __syncthreads();
  }

  // fused affine epilogue + deterministic in-block logdet reduction
#pragma unroll
  for (int mf = 0; mf < MF; ++mf) {
    const int rowInB0 = wm * WM + mf * 16 + ((lane >> 4) << 3);
#pragma unroll
    for (int r = 0; r < 8; ++r) {
      const int rowInB = rowInB0 + r;
      const size_t row = (size_t)mb * BM + rowInB;
      float lsum = 0.f;
#pragma unroll
      for (int nf = 0; nf < NF; ++nf) {
        const int j = wn * WN + nf * 16 + r16;
        float s  = tanhf(acc_s[mf][nf][r] + bs[j]);
        float tv = acc_t[mf][nf][r] + bt[j];
        float2 xe = *(const float2*)(x + row * D_ + 2 * j);
        float2 o;
        o.x = xe.x;
        o.y = xe.y * __expf(s) + tv;
        *(float2*)(out + row * D_ + 2 * j) = o;
        lsum += s;
      }
#pragma unroll
      for (int off = 1; off < 16; off <<= 1) lsum += __shfl_xor(lsum, off, 32);
      if (r16 == 0) red[rowInB][wn] = lsum;
    }
  }
  __syncthreads();
  if (tid < BM) logdet[(size_t)mb * BM + tid] = red[tid][0] + red[tid][1];
}

// ---------------- launch ----------------
extern "C" void kernel_launch(void* const* d_in, const int* in_sizes, int n_in,
                              void* d_out, int out_size, void* d_ws, size_t ws_size,
                              hipStream_t stream) {
  (void)in_sizes; (void)n_in; (void)out_size; (void)ws_size;
  const float* x    = (const float*)d_in[0];
  const float* cond = (const float*)d_in[1];
  const float* W1   = (const float*)d_in[2];
  const float* b1   = (const float*)d_in[3];
  const float* W2   = (const float*)d_in[4];
  const float* b2   = (const float*)d_in[5];
  const float* Ws   = (const float*)d_in[6];
  const float* bs   = (const float*)d_in[7];
  const float* Wt   = (const float*)d_in[8];
  const float* bt   = (const float*)d_in[9];

  float* out    = (float*)d_out;
  float* logdet = out + (size_t)B_ * D_;

  bf16_t* h1  = (bf16_t*)d_ws;                       // B x H
  bf16_t* h2  = h1  + (size_t)B_ * H_;               // B x H
  bf16_t* A0  = h2  + (size_t)B_ * H_;               // B x NETIN
  bf16_t* W1t = A0  + (size_t)B_ * NETIN_;           // H x NETIN
  bf16_t* W2t = W1t + (size_t)H_ * NETIN_;           // H x H
  bf16_t* Wst = W2t + (size_t)H_ * H_;               // SPLIT x H
  bf16_t* Wtt = Wst + (size_t)SPLIT_ * H_;           // SPLIT x H

  pack_a0<<<(B_ * NETIN_) / 256, 256, 0, stream>>>(x, cond, A0);
  convert_transpose<<<(NETIN_ * H_ + 255) / 256, 256, 0, stream>>>(W1, W1t, NETIN_, H_);
  convert_transpose<<<(H_ * H_ + 255) / 256, 256, 0, stream>>>(W2, W2t, H_, H_);
  convert_transpose<<<(H_ * SPLIT_ + 255) / 256, 256, 0, stream>>>(Ws, Wst, H_, SPLIT_);
  convert_transpose<<<(H_ * SPLIT_ + 255) / 256, 256, 0, stream>>>(Wt, Wtt, H_, SPLIT_);

  dim3 g1(H_ / BN, B_ / BM);
  gemm_bf16_relu<<<g1, 256, 0, stream>>>(A0, W1t, b1, h1, NETIN_, H_);
  gemm_bf16_relu<<<g1, 256, 0, stream>>>(h1, W2t, b2, h2, H_, H_);
  gemm3_affine<<<B_ / BM, 256, 0, stream>>>(h2, Wst, Wtt, bs, bt, x, out, logdet);
}
